// GRUModel_41343355191699
// MI455X (gfx1250) — matrix operand order repaired and checked by reading, not compiled, not run on hardware
//
#include <hip/hip_runtime.h>

// GRU (2-layer, B=64,T=512,IN=256,H=1024,OUT=1) for MI455X / gfx1250.
// bf16 WMMA (v_wmma_f32_16x16x32_bf16), f32 accumulate.
// Recurrent scans run as PERSISTENT kernels (32 co-resident blocks) with a
// software grid barrier per timestep -> 11 total kernel launches.
// Workspace required ~485 MB.

#define BSZ  64
#define TSZ  512
#define INSZ 256
#define HSZ  1024
#define G3   (3 * HSZ)
#define NBLK 32   // persistent scan blocks (4 M-tiles x 8 column blocks)

typedef __bf16 v16bf __attribute__((ext_vector_type(16)));
typedef __bf16 v8bf  __attribute__((ext_vector_type(8)));
typedef float  v8f   __attribute__((ext_vector_type(8)));

static __device__ inline v16bf concat8(v8bf lo, v8bf hi) {
    return __builtin_shufflevector(lo, hi, 0,1,2,3,4,5,6,7,8,9,10,11,12,13,14,15);
}
// B-fragment: lane holds column (lane&15), K-runs {8*(lane>>4) ..}+{16+8*(lane>>4) ..}
// from a row-major W row (W rows == B columns of h*W^T).
static __device__ inline v16bf load_bfrag(const __bf16* __restrict__ p) {
    return concat8(*(const v8bf*)p, *(const v8bf*)(p + 16));
}
// A-fragment: lane m = lane&15, K-runs {kb+8*hi ..}+{kb+16+8*hi ..}.
static __device__ inline v16bf load_afrag(const __bf16* row, int kb, int hi) {
    return concat8(*(const v8bf*)(row + kb + 8 * hi),
                   *(const v8bf*)(row + kb + 16 + 8 * hi));
}
static __device__ inline float sigm_(float x) { return 1.f / (1.f + __expf(-x)); }
static __device__ inline float tanh_(float x) { return 2.f / (1.f + __expf(-2.f * x)) - 1.f; }

#define WMMA_BF16(A, Bm, C) \
    __builtin_amdgcn_wmma_f32_16x16x32_bf16(false, (A), false, (Bm), (short)0, (C), false, false)

// ---------------------------------------------------------------- weight cvt
__global__ void k_cvt(const float* __restrict__ src, __bf16* __restrict__ dst, int n) {
    for (int i = blockIdx.x * blockDim.x + threadIdx.x; i < n; i += gridDim.x * blockDim.x)
        dst[i] = (__bf16)src[i];
}

__global__ void k_zero(float* __restrict__ f, int nf, __bf16* __restrict__ b, int nb,
                       unsigned* __restrict__ bar, int nbar) {
    int i = blockIdx.x * blockDim.x + threadIdx.x;
    int stride = gridDim.x * blockDim.x;
    for (int j = i; j < nf; j += stride) f[j] = 0.f;
    for (int j = i; j < nb; j += stride) b[j] = (__bf16)0.f;
    for (int j = i; j < nbar; j += stride) bar[j] = 0u;
}

// ------------------------------------------------- input projection (GEMM)
// xg[m][g] = sum_k A[m][k] * Wih[g][k] + bih[g], m = t*B + b (time-major out).
// Layer 0: A = x fp32 [B,T,IN] with row remap; Layer 1: A = y bf16 [T*B,H].
// One wave -> 16x64 output tile (A fragment reused across 4 B tiles).
template <bool A_IS_F32>
__global__ __launch_bounds__(256) void k_inproj(const void* __restrict__ Ain,
                                                const __bf16* __restrict__ Wb,
                                                const float* __restrict__ bias,
                                                float* __restrict__ xg,
                                                int M, int N, int K) {
    int wid  = (blockIdx.x * blockDim.x + threadIdx.x) >> 5;
    int lane = threadIdx.x & 31;
    int nWavesN = N / 64;
    int mi = wid / nWavesN;
    int ni = wid - mi * nWavesN;
    if (mi * 16 >= M) return;
    int m0 = mi * 16, n0 = ni * 64;
    int nl = lane & 15, hi = lane >> 4;
    int mrow = m0 + nl;

    const float*  arowf = nullptr;
    const __bf16* arowb = nullptr;
    if (A_IS_F32) {
        int b = mrow & (BSZ - 1);
        int t = mrow >> 6;  // B == 64
        arowf = (const float*)Ain + ((size_t)b * TSZ + t) * K;
    } else {
        arowb = (const __bf16*)Ain + (size_t)mrow * K;
    }

    v8f acc[4] = {};
    #pragma unroll 2
    for (int kb = 0; kb < K; kb += 32) {
        v16bf af;
        if (A_IS_F32) {
            v8f f0 = *(const v8f*)(arowf + kb + 8 * hi);
            v8f f1 = *(const v8f*)(arowf + kb + 16 + 8 * hi);
            v16bf t;
            #pragma unroll
            for (int i = 0; i < 8; ++i) { t[i] = (__bf16)f0[i]; t[i + 8] = (__bf16)f1[i]; }
            af = t;
        } else {
            af = load_afrag(arowb, kb, hi);
        }
        #pragma unroll
        for (int g = 0; g < 4; ++g) {
            const __bf16* bp = Wb + (size_t)(n0 + g * 16 + nl) * K + kb + 8 * hi;
            v16bf bf = load_bfrag(bp);
            acc[g] = WMMA_BF16(af, bf, acc[g]);
        }
    }
    #pragma unroll
    for (int g = 0; g < 4; ++g) {
        int col = n0 + g * 16 + nl;
        float bv = bias[col];
        #pragma unroll
        for (int r = 0; r < 8; ++r) {
            int row = m0 + r + 8 * hi;
            xg[(size_t)row * N + col] = acc[g][r] + bv;
        }
    }
}

// ------------------------------------------------- persistent recurrent scan
// 32 co-resident blocks loop over all T timesteps; per step each block stages
// its 16 h-rows (bf16) into LDS, each of its 8 waves owns one 16-column gate
// group and computes r/z/n tiles fused. K loop is ping-pong unrolled x2: each
// fragment set is REDEFINED BY A LOAD (never register-copied), so the backend
// keeps the pipeline without v_mov rotation or hazard nops. Steps separated
// by a software grid barrier over ping-pong h buffers in L2.
__global__ __launch_bounds__(256) void k_scan(const __bf16* __restrict__ Whh,
                                              const float* __restrict__ bhh,
                                              const float* __restrict__ xg,
                                              float* __restrict__ hf,   // [2][B][H]
                                              __bf16* __restrict__ hb,  // [2][B][H]
                                              __bf16* __restrict__ y,   // [T][B][H] or scratch
                                              int store_y,
                                              unsigned* __restrict__ bar) {  // bar[0]=arrive, bar[1]=gen
    __shared__ __align__(16) __bf16 sA[16 * HSZ];  // 32 KB
    const int tid  = threadIdx.x;
    const int mi   = blockIdx.x >> 3;
    const int lane = tid & 31;
    const int w    = tid >> 5;
    const int nl   = lane & 15, hi = lane >> 4;
    const int jb   = ((blockIdx.x & 7) * 8 + w) * 16;
    const int j    = jb + nl;

    const __bf16* br = Whh + (size_t)(jb + nl) * HSZ + 8 * hi;
    const __bf16* bz = Whh + (size_t)(HSZ + jb + nl) * HSZ + 8 * hi;
    const __bf16* bn = Whh + (size_t)(2 * HSZ + jb + nl) * HSZ + 8 * hi;
    const float bhr = bhh[j], bhz = bhh[HSZ + j], bhn = bhh[2 * HSZ + j];

    for (int t = 0; t < TSZ; ++t) {
        const int pi = t & 1, po = pi ^ 1;
        const float*  xgt   = xg + (size_t)t * BSZ * G3;
        const float*  hin_f = hf + (size_t)pi * BSZ * HSZ;
        const __bf16* hin_b = hb + (size_t)pi * BSZ * HSZ;
        float*  hout_f = hf + (size_t)po * BSZ * HSZ;
        __bf16* hout_b = hb + (size_t)po * BSZ * HSZ;

        // stage this block's 16 h rows into LDS
        {
            const uint4* src = (const uint4*)(hin_b + (size_t)mi * 16 * HSZ);
            uint4* dst = (uint4*)sA;
            #pragma unroll
            for (int i = tid; i < (16 * HSZ * 2) / 16; i += 256) dst[i] = src[i];
        }
        __syncthreads();

        const __bf16* sArow = sA + (size_t)nl * HSZ;
        v8f cr = {}, cz = {}, cn = {};

        // prologue: fill both fragment sets (k = 0 and k = 32)
        v16bf afA = load_afrag(sArow, 0, hi);
        v16bf frA = load_bfrag(br);
        v16bf fzA = load_bfrag(bz);
        v16bf fnA = load_bfrag(bn);
        v16bf afB = load_afrag(sArow, 32, hi);
        v16bf frB = load_bfrag(br + 32);
        v16bf fzB = load_bfrag(bz + 32);
        v16bf fnB = load_bfrag(bn + 32);

        #pragma unroll 1
        for (int kb = 0; kb + 64 < HSZ; kb += 64) {
            // consume set A (kb), refill set A (kb+64)
            cr = WMMA_BF16(afA, frA, cr);
            cz = WMMA_BF16(afA, fzA, cz);
            cn = WMMA_BF16(afA, fnA, cn);
            afA = load_afrag(sArow, kb + 64, hi);
            frA = load_bfrag(br + kb + 64);
            fzA = load_bfrag(bz + kb + 64);
            fnA = load_bfrag(bn + kb + 64);
            // consume set B (kb+32), refill set B (kb+96)
            cr = WMMA_BF16(afB, frB, cr);
            cz = WMMA_BF16(afB, fzB, cz);
            cn = WMMA_BF16(afB, fnB, cn);
            afB = load_afrag(sArow, kb + 96, hi);
            frB = load_bfrag(br + kb + 96);
            fzB = load_bfrag(bz + kb + 96);
            fnB = load_bfrag(bn + kb + 96);
        }
        // epilogue: last two k-chunks (HSZ-64, HSZ-32)
        cr = WMMA_BF16(afA, frA, cr);
        cz = WMMA_BF16(afA, fzA, cz);
        cn = WMMA_BF16(afA, fnA, cn);
        cr = WMMA_BF16(afB, frB, cr);
        cz = WMMA_BF16(afB, fzB, cz);
        cn = WMMA_BF16(afB, fnB, cn);

        // fused GRU gate math + h update
        #pragma unroll
        for (int r = 0; r < 8; ++r) {
            int b = mi * 16 + r + 8 * hi;
            const float* xrow = xgt + (size_t)b * G3;
            float xr = xrow[j], xz = xrow[HSZ + j], xn = xrow[2 * HSZ + j];
            float rg = sigm_(xr + cr[r] + bhr);
            float zg = sigm_(xz + cz[r] + bhz);
            float ng = tanh_(xn + rg * (cn[r] + bhn));
            float hp = hin_f[(size_t)b * HSZ + j];
            float hv = (1.f - zg) * ng + zg * hp;
            hout_f[(size_t)b * HSZ + j] = hv;
            hout_b[(size_t)b * HSZ + j] = (__bf16)hv;
            if (store_y) y[(size_t)t * BSZ * HSZ + (size_t)b * HSZ + j] = (__bf16)hv;
        }

        // ---- grid barrier between timesteps ----
        __threadfence();   // each wave flushes its own h stores device-scope
        __syncthreads();
        if (tid == 0)
            if (__hip_atomic_fetch_add(&bar[0], 1u, __ATOMIC_ACQ_REL,
                                       __HIP_MEMORY_SCOPE_AGENT) == (unsigned)(NBLK * (t + 1) - 1))
                __hip_atomic_fetch_add(&bar[1], 1u, __ATOMIC_RELEASE, __HIP_MEMORY_SCOPE_AGENT);
        if (lane == 0)     // one spinner per wave: per-wave acquire -> cache inv
            while (__hip_atomic_load(&bar[1], __ATOMIC_ACQUIRE,
                                     __HIP_MEMORY_SCOPE_AGENT) < (unsigned)(t + 1))
                __builtin_amdgcn_s_sleep(1);
        __syncthreads();
    }
}

// ------------------------------------------------------------------- final FC
__global__ void k_fc(const float* __restrict__ hf, const float* __restrict__ Wfc,
                     const float* __restrict__ bfc, float* __restrict__ out) {
    int b = threadIdx.x;
    if (b < BSZ) {
        float s = 0.f;
        for (int jj = 0; jj < HSZ; ++jj) s += hf[(size_t)b * HSZ + jj] * Wfc[jj];
        out[b] = s + bfc[0];
    }
}

extern "C" void kernel_launch(void* const* d_in, const int* in_sizes, int n_in,
                              void* d_out, int out_size, void* d_ws, size_t ws_size,
                              hipStream_t stream) {
    const float* x    = (const float*)d_in[0];
    const float* Wih0 = (const float*)d_in[1];
    const float* Whh0 = (const float*)d_in[2];
    const float* bih0 = (const float*)d_in[3];
    const float* bhh0 = (const float*)d_in[4];
    const float* Wih1 = (const float*)d_in[5];
    const float* Whh1 = (const float*)d_in[6];
    const float* bih1 = (const float*)d_in[7];
    const float* bhh1 = (const float*)d_in[8];
    const float* Wfc  = (const float*)d_in[9];
    const float* bfc  = (const float*)d_in[10];
    float* out = (float*)d_out;

    char* ws = (char*)d_ws;
    size_t off = 0;
    auto alloc = [&](size_t bytes) -> char* {
        char* p = ws + off;
        off = (off + bytes + 255) & ~(size_t)255;
        return p;
    };
    __bf16*   wih0b = (__bf16*)alloc((size_t)G3 * INSZ * 2);
    __bf16*   whh0b = (__bf16*)alloc((size_t)G3 * HSZ * 2);
    __bf16*   wih1b = (__bf16*)alloc((size_t)G3 * HSZ * 2);
    __bf16*   whh1b = (__bf16*)alloc((size_t)G3 * HSZ * 2);
    float*    xg    = (float*)alloc((size_t)TSZ * BSZ * G3 * 4);
    __bf16*   y0    = (__bf16*)alloc((size_t)TSZ * BSZ * HSZ * 2);
    float*    hf    = (float*)alloc((size_t)2 * BSZ * HSZ * 4);
    __bf16*   hb    = (__bf16*)alloc((size_t)2 * BSZ * HSZ * 2);
    unsigned* bar0  = (unsigned*)alloc(2 * sizeof(unsigned));
    unsigned* bar1  = (unsigned*)alloc(2 * sizeof(unsigned));

    k_cvt<<<256, 256, 0, stream>>>(Wih0, wih0b, G3 * INSZ);
    k_cvt<<<256, 256, 0, stream>>>(Whh0, whh0b, G3 * HSZ);
    k_cvt<<<256, 256, 0, stream>>>(Wih1, wih1b, G3 * HSZ);
    k_cvt<<<256, 256, 0, stream>>>(Whh1, whh1b, G3 * HSZ);

    const int M = BSZ * TSZ;                       // 32768
    const int nBlocks = (M / 16) * (G3 / 64) / 8;  // 8 waves/block

    // ---- layer 0 ----
    k_inproj<true><<<nBlocks, 256, 0, stream>>>((const void*)x, wih0b, bih0, xg, M, G3, INSZ);
    k_zero<<<64, 256, 0, stream>>>(hf, 2 * BSZ * HSZ, hb, 2 * BSZ * HSZ, bar0, 2);
    k_scan<<<NBLK, 256, 0, stream>>>(whh0b, bhh0, xg, hf, hb, y0, 1, bar0);
    // ---- layer 1 ----
    k_inproj<false><<<nBlocks, 256, 0, stream>>>((const void*)y0, wih1b, bih1, xg, M, G3, HSZ);
    k_zero<<<64, 256, 0, stream>>>(hf, 2 * BSZ * HSZ, hb, 2 * BSZ * HSZ, bar1, 2);
    k_scan<<<NBLK, 256, 0, stream>>>(whh1b, bhh1, xg, hf, hb, y0, 0, bar1);
    // final hidden state is in buffer 0 (t=511: in=1, out=0)
    k_fc<<<1, 64, 0, stream>>>(hf, Wfc, bfc, out);
}
